// ProductQuantize_89928025243835
// MI455X (gfx1250) — matrix operand order repaired and checked by reading, not compiled, not run on hardware
//
#include <hip/hip_runtime.h>

#define B_TOT 8192
#define M_TOT 4
#define K_TOT 2048
#define D_TOT 256
#define TILE_B 128
#define KCHUNK 128

typedef __attribute__((ext_vector_type(16))) __bf16        v16bf;
typedef __attribute__((ext_vector_type(8)))  float         v8f;
typedef __attribute__((ext_vector_type(4)))  float         f32x4;
typedef __attribute__((ext_vector_type(4)))  unsigned int  u32x4;
typedef __attribute__((ext_vector_type(2)))  unsigned int  u32x2;

union ABRegs { v16bf v; u32x4 q[2]; };

// fp32 -> bf16 round-to-nearest-even, packed pair
__device__ __forceinline__ unsigned pack_bf16x2(float lo, float hi) {
  unsigned ul = __float_as_uint(lo);
  unsigned uh = __float_as_uint(hi);
  ul = (ul + 0x7FFFu + ((ul >> 16) & 1u)) >> 16;
  uh = (uh + 0x7FFFu + ((uh >> 16) & 1u)) >> 16;
  return ul | (uh << 16);
}

// Precompute |w|^2 for all M*K codewords (one wave per row).
__global__ __launch_bounds__(256)
void pq_w2_kernel(const float* __restrict__ codebooks, float* __restrict__ w2) {
  const int row  = blockIdx.x * 8 + (threadIdx.x >> 5);
  const int lane = threadIdx.x & 31;
  const float* src = codebooks + (size_t)row * D_TOT;
  float s = 0.f;
  #pragma unroll
  for (int d = lane; d < D_TOT; d += 32) { float v = src[d]; s = fmaf(v, v, s); }
  #pragma unroll
  for (int off = 16; off >= 1; off >>= 1) s += __shfl_xor(s, off, 32);
  if (lane == 0) w2[row] = s;
}

__global__ __launch_bounds__(256, 1)
void pq_main_kernel(const float* __restrict__ patches,
                    const float* __restrict__ codebooks,
                    const float* __restrict__ w2,
                    float* __restrict__ emb,
                    float* __restrict__ codes,
                    float* __restrict__ lossAcc)
{
  __shared__ unsigned short lA[TILE_B * D_TOT];   // 64 KB: patches tile, bf16
  __shared__ unsigned short lB[KCHUNK * D_TOT];   // 64 KB: codebook chunk, bf16
  __shared__ int lCodes[TILE_B];

  const int tid  = threadIdx.x;
  const int lane = tid & 31;
  const int wv   = tid >> 5;                 // wave 0..7, owns rows [16wv,16wv+16)
  const int b0   = blockIdx.x * TILE_B;
  const int m    = blockIdx.y;

  const float* cb = codebooks + (size_t)m * K_TOT * D_TOT;

  // Stage patch tile to LDS as bf16 (float4 loads, packed bf16x2 stores).
  for (int i = tid; i < TILE_B * (D_TOT / 4); i += 256) {
    const int r  = i >> 6;                   // D_TOT/4 == 64 float4 per row
    const int c4 = i & 63;
    f32x4 v = *(const f32x4*)&patches[((size_t)(b0 + r) * M_TOT + m) * D_TOT + c4 * 4];
    u32x2 p;
    p.x = pack_bf16x2(v.x, v.y);
    p.y = pack_bf16x2(v.z, v.w);
    *(u32x2*)&lA[r * D_TOT + c4 * 4] = p;
  }
  __syncthreads();

  // ISA 7.12.2 bf16 operand layouts (wave32):
  //  A 16x32: lanes 0-15 -> M=lane, halves K{0..7,16..23}; lanes 16-31 -> K{8..15,24..31}
  //  B 32x16: lanes 0-15 -> N=lane, halves K{0..7,16..23}; lanes 16-31 -> K{8..15,24..31}
  const int rowA  = 16 * wv + (lane & 15);
  const int ahalf = (lane >= 16) ? 8 : 0;
  const int bhalf = (lane >= 16) ? 8 : 0;

  // A operands are invariant across all K-chunks/column-tiles: hoist all 8
  // dk-slices (64 VGPRs) out of the hot loop so each WMMA only waits on B.
  ABRegs aR[8];
  #pragma unroll
  for (int dk = 0; dk < D_TOT / 32; ++dk) {
    const int dbase = dk * 32;
    aR[dk].q[0] = *(const u32x4*)&lA[rowA * D_TOT + dbase + ahalf];
    aR[dk].q[1] = *(const u32x4*)&lA[rowA * D_TOT + dbase + ahalf + 16];
  }

  float minv[8]; int mini[8];
  #pragma unroll
  for (int i = 0; i < 8; ++i) { minv[i] = 3.4e38f; mini[i] = 0; }

  for (int chunk = 0; chunk < K_TOT / KCHUNK; ++chunk) {
    __syncthreads();
    for (int i = tid; i < KCHUNK * (D_TOT / 4); i += 256) {
      const int r  = i >> 6;
      const int c4 = i & 63;
      f32x4 v = *(const f32x4*)&cb[((size_t)(chunk * KCHUNK + r)) * D_TOT + c4 * 4];
      u32x2 p;
      p.x = pack_bf16x2(v.x, v.y);
      p.y = pack_bf16x2(v.z, v.w);
      *(u32x2*)&lB[r * D_TOT + c4 * 4] = p;
    }
    __syncthreads();

    // Overlap next chunk's L2 fill with this chunk's WMMAs (global_prefetch_b8).
    if (chunk + 1 < K_TOT / KCHUNK) {
      const float* nxt = &cb[(size_t)((chunk + 1) * KCHUNK) * D_TOT];
      #pragma unroll
      for (int i = 0; i < 4; ++i)
        __builtin_prefetch(&nxt[(tid + i * 256) * 32], 0, 1);
    }

    #pragma unroll
    for (int ct = 0; ct < KCHUNK / 16; ++ct) {
      const int klocal = ct * 16 + (lane & 15);
      v8f acc = {};
      #pragma unroll
      for (int dk = 0; dk < D_TOT / 32; ++dk) {
        const int dbase = dk * 32;
        ABRegs b;
        b.q[0] = *(const u32x4*)&lB[klocal * D_TOT + dbase + bhalf];
        b.q[1] = *(const u32x4*)&lB[klocal * D_TOT + dbase + bhalf + 16];
        acc = __builtin_amdgcn_wmma_f32_16x16x32_bf16(false, aR[dk].v, false, b.v,
                                                      (short)0, acc, false, false);
      }
      // score = |w|^2 - 2 x.w  (argmin-equivalent to the L2 distance)
      const int   kglob = chunk * KCHUNK + klocal;
      const float w2v   = w2[m * K_TOT + kglob];
      #pragma unroll
      for (int i = 0; i < 8; ++i) {
        const float s = fmaf(-2.0f, acc[i], w2v);
        if (s < minv[i]) { minv[i] = s; mini[i] = kglob; }
      }
    }
  }

  // Argmin reduction within each 16-lane half (C-layout: VGPR i -> row i / i+8).
  #pragma unroll
  for (int i = 0; i < 8; ++i) {
    #pragma unroll
    for (int off = 8; off >= 1; off >>= 1) {
      const float ov = __shfl_xor(minv[i], off, 32);
      const int   oi = __shfl_xor(mini[i], off, 32);
      if (ov < minv[i] || (ov == minv[i] && oi < mini[i])) { minv[i] = ov; mini[i] = oi; }
    }
    if ((lane & 15) == 0)
      lCodes[16 * wv + i + ((lane >= 16) ? 8 : 0)] = mini[i];
  }
  __syncthreads();

  // Gather winning codeword in fp32, emit embeddings = p + (q - p), accumulate MSE.
  float sumsq = 0.f;
  for (int e = tid; e < TILE_B * D_TOT; e += 256) {
    const int r = e >> 8;
    const int d = e & 255;
    const int c = lCodes[r];
    const size_t gidx = ((size_t)(b0 + r) * M_TOT + m) * D_TOT + d;
    const float q = cb[(size_t)c * D_TOT + d];
    const float p = patches[gidx];
    const float diff = q - p;
    emb[gidx] = p + diff;
    sumsq = fmaf(diff, diff, sumsq);
  }
  #pragma unroll
  for (int off = 16; off >= 1; off >>= 1) sumsq += __shfl_xor(sumsq, off, 32);
  if (lane == 0) atomicAdd(lossAcc, sumsq);

  if (tid < TILE_B)
    codes[(size_t)(b0 + tid) * M_TOT + m] = (float)lCodes[tid];
}

__global__ void pq_finalize_kernel(const float* __restrict__ acc,
                                   float* __restrict__ loss) {
  // loss = codebook_loss + 0.25 * commitment_loss = 1.25 * MSE
  *loss = 1.25f * acc[0] / (float)((size_t)B_TOT * M_TOT * D_TOT);
}

extern "C" void kernel_launch(void* const* d_in, const int* in_sizes, int n_in,
                              void* d_out, int out_size, void* d_ws, size_t ws_size,
                              hipStream_t stream) {
  const float* patches   = (const float*)d_in[0];
  const float* codebooks = (const float*)d_in[1];

  float* out   = (float*)d_out;
  float* emb   = out;                                        // [B,M,D]
  float* codes = emb + (size_t)B_TOT * M_TOT * D_TOT;        // [B,M]
  float* loss  = codes + (size_t)B_TOT * M_TOT;              // [1]

  float* acc = (float*)d_ws;                                 // loss accumulator
  float* w2  = acc + 16;                                     // [M*K] squared norms

  hipMemsetAsync(d_ws, 0, 64, stream);
  pq_w2_kernel<<<dim3((M_TOT * K_TOT) / 8), 256, 0, stream>>>(codebooks, w2);
  pq_main_kernel<<<dim3(B_TOT / TILE_B, M_TOT), 256, 0, stream>>>(
      patches, codebooks, w2, emb, codes, acc);
  pq_finalize_kernel<<<1, 1, 0, stream>>>(acc, loss);
}
